// HierarchicalAttention_30958124270368
// MI455X (gfx1250) — matrix operand order repaired and checked by reading, not compiled, not run on hardware
//
#include <hip/hip_runtime.h>
#include <hip/hip_bf16.h>
#include <math.h>

// ---------------------------------------------------------------------------
// HierarchicalAttention on gfx1250 (CDNA5, wave32, WMMA)
//   N=4096 nodes, E=131072 edges, HIDDEN=256, HEADS=8, HEAD_DIM=32
// Dense NxN-per-head attention collapses algebraically (exp(0)=1 for
// non-edges) to: rank-1 sumV term + sparse edge scatter, normalized by
// N + sum(e^s - 1).  GEMMs run as bf16 WMMA 16x16x32 with f32 accum,
// register-blocked 4 column-tiles per wave (16x64 strip) for 4 WMMAs per
// A-fragment load.
// ---------------------------------------------------------------------------

typedef unsigned short ushort_t;
typedef unsigned int   uint_t;
typedef __attribute__((ext_vector_type(16))) __bf16 bf16x16;
typedef __attribute__((ext_vector_type(8)))  float  f32x8;
typedef __attribute__((ext_vector_type(4)))  uint_t u32x4;

#define N_NODES 4096
#define N_EDGES 131072
#define HIDDEN  256
#define HEADS   8
#define HEAD_DIM 32
#define INV_SCALE 0.17677669529663687f   // 1/sqrt(32)

union FragBF { u32x4 q[2]; bf16x16 v; };

__device__ __forceinline__ ushort_t f2bf(float f) {
  uint_t u = __float_as_uint(f);
  u += 0x7FFFu + ((u >> 16) & 1u);     // round-to-nearest-even
  return (ushort_t)(u >> 16);
}

__device__ __forceinline__ float wave_reduce_sum(float v) {
  #pragma unroll
  for (int off = 16; off > 0; off >>= 1) v += __shfl_xor(v, off, 32);
  return v;
}

// ---------------------------- packing / zero ------------------------------
__global__ void pack_bf16_kernel(const float* __restrict__ src,
                                 ushort_t* __restrict__ dst, int n) {
  int i = blockIdx.x * blockDim.x + threadIdx.x;
  if (i < n) dst[i] = f2bf(src[i]);
}

__global__ void zero_f32_kernel(float* __restrict__ p, int n) {
  int i = blockIdx.x * blockDim.x + threadIdx.x;
  if (i < n) p[i] = 0.0f;
}

// ---------------------- WMMA GEMM: out = A @ W^T + b ----------------------
// A: [M,K] bf16 row-major, W: [N,K] bf16 row-major, out: [M,N] f32.
// One wave per 16x64 output strip (4 column tiles register-blocked).
// Fragment layouts per ISA 7.12.2:
//   A (16x32 bf16): lane holds row m=lane&15; K-halves select by lane>>4.
//   B (32x16 bf16): lane holds col n=lane&15; same k order as A (8-K chunks at 8*half and 16+8*half).
#define NBLK 4
__global__ void gemm_wmma_bf16_kernel(const ushort_t* __restrict__ A,
                                      const ushort_t* __restrict__ W,
                                      const float* __restrict__ bias,
                                      float* __restrict__ out,
                                      int M, int N, int K, int relu) {
  const int gw   = (blockIdx.x * blockDim.x + threadIdx.x) >> 5;
  const int lane = threadIdx.x & 31;
  const int strips_n = N >> 6;                   // 64-wide strips
  const int strips   = (M >> 4) * strips_n;
  if (gw >= strips) return;
  const int tm = gw / strips_n, tn = gw % strips_n;
  const int row0 = tm << 4, col0 = tn << 6;
  const int mn   = lane & 15;       // A row within tile == B col within tile
  const int half = lane >> 4;

  const ushort_t* pa = A + (size_t)(row0 + mn) * K + half * 8;   // kb = half*8
  const ushort_t* pb0 = W + (size_t)(col0 + mn) * K + half * 8;  // kc = half*8

  FragBF a, b;
  f32x8 c[NBLK];
  #pragma unroll
  for (int j = 0; j < NBLK; ++j) c[j] = (f32x8){};

  for (int k0 = 0; k0 < K; k0 += 32) {
    a.q[0] = *(const u32x4*)(pa + k0);        // k = k0+kb+0..7
    a.q[1] = *(const u32x4*)(pa + k0 + 16);   // k = k0+kb+16..23
    if (k0 + 32 < K) {                        // global_prefetch_b8 next slab
      __builtin_prefetch(pa + k0 + 32, 0, 3);
      __builtin_prefetch(pb0 + k0 + 32, 0, 3);
    }
    #pragma unroll
    for (int j = 0; j < NBLK; ++j) {
      const ushort_t* pb = pb0 + (size_t)(j * 16) * K;
      b.q[0] = *(const u32x4*)(pb + k0);      // k = k0+kc+0..7
      b.q[1] = *(const u32x4*)(pb + k0 + 16); // k = k0+kc+16..23
      c[j] = __builtin_amdgcn_wmma_f32_16x16x32_bf16(
                 /*neg_a=*/false, a.v, /*neg_b=*/false, b.v,
                 /*c_mod=*/(short)0, c[j], /*reuse_a=*/false, /*reuse_b=*/false);
    }
  }

  const int mb = half * 8;                    // D: vgpr v -> row v + half*8
  #pragma unroll
  for (int j = 0; j < NBLK; ++j) {
    const int col = col0 + j * 16 + mn;
    const float bv = bias[col];
    #pragma unroll
    for (int v = 0; v < 8; ++v) {
      float r = c[j][v] + bv;
      if (relu) r = fmaxf(r, 0.0f);
      out[(size_t)(row0 + mb + v) * N + col] = r;
    }
  }
}

// --------------------------- column reductions ----------------------------
// dst[c] = scale * sum_i src[i*cols + c]
__global__ void colsum_kernel(const float* __restrict__ src,
                              float* __restrict__ dst,
                              int rows, int cols, float scale) {
  int c = blockIdx.x * blockDim.x + threadIdx.x;
  if (c >= cols) return;
  float s = 0.0f;
  for (int i = 0; i < rows; ++i) s += src[(size_t)i * cols + c];
  dst[c] = s * scale;
}

// Kg = Wgk @ ctx + bgk ; Vg = Wgv @ ctx + bgv   (one block of 256 threads)
__global__ void graph_vec_kernel(const float* __restrict__ Wgk,
                                 const float* __restrict__ bgk,
                                 const float* __restrict__ Wgv,
                                 const float* __restrict__ bgv,
                                 const float* __restrict__ ctx,
                                 float* __restrict__ Kg,
                                 float* __restrict__ Vg) {
  int o = threadIdx.x;
  float sk = 0.0f, sv = 0.0f;
  for (int k = 0; k < HIDDEN; ++k) {
    float cv = ctx[k];
    sk += Wgk[o * HIDDEN + k] * cv;
    sv += Wgv[o * HIDDEN + k] * cv;
  }
  Kg[o] = sk + bgk[o];
  Vg[o] = sv + bgv[o];
}

// s[i] = (Qg[i,:] . Kg) / sqrt(32)   (one wave per node)
__global__ void node_score_kernel(const float* __restrict__ Qg,
                                  const float* __restrict__ Kg,
                                  float* __restrict__ sraw, int n) {
  int gw   = (blockIdx.x * blockDim.x + threadIdx.x) >> 5;
  int lane = threadIdx.x & 31;
  if (gw >= n) return;
  float p = 0.0f;
  #pragma unroll
  for (int k = lane; k < HIDDEN; k += 32) p += Qg[(size_t)gw * HIDDEN + k] * Kg[k];
  p = wave_reduce_sum(p);
  if (lane == 0) sraw[gw] = p * INV_SCALE;
}

// softmax over all N nodes, single block of 256 threads
__global__ void softmax_kernel(const float* __restrict__ s,
                               float* __restrict__ w, int n) {
  __shared__ float red[256];
  int t = threadIdx.x;
  float m = -3.402823466e+38f;
  for (int i = t; i < n; i += 256) m = fmaxf(m, s[i]);
  red[t] = m; __syncthreads();
  for (int o = 128; o > 0; o >>= 1) {
    if (t < o) red[t] = fmaxf(red[t], red[t + o]);
    __syncthreads();
  }
  float mx = red[0]; __syncthreads();
  float acc = 0.0f;
  for (int i = t; i < n; i += 256) acc += __expf(s[i] - mx);
  red[t] = acc; __syncthreads();
  for (int o = 128; o > 0; o >>= 1) {
    if (t < o) red[t] += red[t + o];
    __syncthreads();
  }
  float inv = 1.0f / red[0];
  for (int i = t; i < n; i += 256) w[i] = __expf(s[i] - mx) * inv;
}

// ----------------------------- edge scatter -------------------------------
// One wave per (edge, head); lane = dim.  g = e^{s} - 1 scattered to both
// endpoints (A + A^T symmetrization).  acc/Z are 4 MB / 128 KB -> L2-resident.
#define EDGES_PER_BLOCK 32
__global__ void edge_scatter_kernel(const int* __restrict__ ei,
                                    const float* __restrict__ Q,
                                    const float* __restrict__ K,
                                    const float* __restrict__ V,
                                    float* __restrict__ acc,
                                    float* __restrict__ Z) {
  const int h = threadIdx.x >> 5;     // 0..7 : head per wave
  const int d = threadIdx.x & 31;     // dim within head
  const int c = h * HEAD_DIM + d;
  const int base = blockIdx.x * EDGES_PER_BLOCK;
  for (int eo = 0; eo < EDGES_PER_BLOCK; ++eo) {
    const int e = base + eo;
    if (e >= N_EDGES) return;
    const int src = ei[e];
    const int tgt = ei[N_EDGES + e];
    const float qs = Q[(size_t)src * HIDDEN + c];
    const float kt = K[(size_t)tgt * HIDDEN + c];
    const float s  = wave_reduce_sum(qs * kt) * INV_SCALE;
    const float g  = __expf(s) - 1.0f;
    atomicAdd(&acc[(size_t)src * HIDDEN + c], g * V[(size_t)tgt * HIDDEN + c]);
    atomicAdd(&acc[(size_t)tgt * HIDDEN + c], g * V[(size_t)src * HIDDEN + c]);
    if (d == 0) {
      atomicAdd(&Z[src * HEADS + h], g);
      atomicAdd(&Z[tgt * HEADS + h], g);
    }
  }
}

// -------- combine: attended + graph_attended -> bf16 [N, 512] -------------
__global__ void combine_kernel(const float* __restrict__ acc,
                               const float* __restrict__ Z,
                               const float* __restrict__ sumV,
                               const float* __restrict__ w,
                               const float* __restrict__ Vg,
                               ushort_t* __restrict__ comb) {
  const int i = blockIdx.x;
  const int c = threadIdx.x;
  const float att = (sumV[c] + acc[(size_t)i * HIDDEN + c]) /
                    ((float)N_NODES + Z[i * HEADS + (c >> 5)]);
  comb[(size_t)i * (2 * HIDDEN) + c]          = f2bf(att);
  comb[(size_t)i * (2 * HIDDEN) + HIDDEN + c] = f2bf(w[i] * Vg[c]);
}

// ---------------------------------------------------------------------------
extern "C" void kernel_launch(void* const* d_in, const int* in_sizes, int n_in,
                              void* d_out, int out_size, void* d_ws, size_t ws_size,
                              hipStream_t stream) {
  const float* x   = (const float*)d_in[0];
  const int*   ei  = (const int*)  d_in[1];
  const float* Wq  = (const float*)d_in[2];  const float* bq  = (const float*)d_in[3];
  const float* Wk  = (const float*)d_in[4];  const float* bk  = (const float*)d_in[5];
  const float* Wv  = (const float*)d_in[6];  const float* bv  = (const float*)d_in[7];
  const float* Wgq = (const float*)d_in[8];  const float* bgq = (const float*)d_in[9];
  const float* Wgk = (const float*)d_in[10]; const float* bgk = (const float*)d_in[11];
  const float* Wgv = (const float*)d_in[12]; const float* bgv = (const float*)d_in[13];
  const float* Wf  = (const float*)d_in[14]; const float* bf  = (const float*)d_in[15];
  float* out = (float*)d_out;

  // ------------- workspace layout (256B aligned) -------------
  char* wsp = (char*)d_ws;
  auto alloc = [&](size_t bytes) -> void* {
    void* p = (void*)wsp;
    wsp += (bytes + 255) & ~(size_t)255;
    return p;
  };
  ushort_t* xh   = (ushort_t*)alloc((size_t)N_NODES * HIDDEN * 2);
  ushort_t* Wqh  = (ushort_t*)alloc((size_t)HIDDEN * HIDDEN * 2);
  ushort_t* Wkh  = (ushort_t*)alloc((size_t)HIDDEN * HIDDEN * 2);
  ushort_t* Wvh  = (ushort_t*)alloc((size_t)HIDDEN * HIDDEN * 2);
  ushort_t* Wgqh = (ushort_t*)alloc((size_t)HIDDEN * HIDDEN * 2);
  ushort_t* Wfh  = (ushort_t*)alloc((size_t)HIDDEN * 2 * HIDDEN * 2);
  float* Q    = (float*)alloc((size_t)N_NODES * HIDDEN * 4);
  float* Kp   = (float*)alloc((size_t)N_NODES * HIDDEN * 4);
  float* V    = (float*)alloc((size_t)N_NODES * HIDDEN * 4);
  float* Qg   = (float*)alloc((size_t)N_NODES * HIDDEN * 4);
  float* acc  = (float*)alloc((size_t)N_NODES * HIDDEN * 4);
  float* Z    = (float*)alloc((size_t)N_NODES * HEADS * 4);
  float* sumV = (float*)alloc(HIDDEN * 4);
  float* ctx  = (float*)alloc(HIDDEN * 4);
  float* Kg   = (float*)alloc(HIDDEN * 4);
  float* Vg   = (float*)alloc(HIDDEN * 4);
  float* sraw = (float*)alloc(N_NODES * 4);
  float* wgt  = (float*)alloc(N_NODES * 4);
  ushort_t* comb = (ushort_t*)alloc((size_t)N_NODES * 2 * HIDDEN * 2);
  (void)ws_size; (void)n_in; (void)in_sizes; (void)out_size;

  const int TB = 256;

  // 1) bf16 packing
  pack_bf16_kernel<<<(N_NODES * HIDDEN + TB - 1) / TB, TB, 0, stream>>>(x, xh, N_NODES * HIDDEN);
  pack_bf16_kernel<<<(HIDDEN * HIDDEN + TB - 1) / TB, TB, 0, stream>>>(Wq,  Wqh,  HIDDEN * HIDDEN);
  pack_bf16_kernel<<<(HIDDEN * HIDDEN + TB - 1) / TB, TB, 0, stream>>>(Wk,  Wkh,  HIDDEN * HIDDEN);
  pack_bf16_kernel<<<(HIDDEN * HIDDEN + TB - 1) / TB, TB, 0, stream>>>(Wv,  Wvh,  HIDDEN * HIDDEN);
  pack_bf16_kernel<<<(HIDDEN * HIDDEN + TB - 1) / TB, TB, 0, stream>>>(Wgq, Wgqh, HIDDEN * HIDDEN);
  pack_bf16_kernel<<<(HIDDEN * 2 * HIDDEN + TB - 1) / TB, TB, 0, stream>>>(Wf, Wfh, HIDDEN * 2 * HIDDEN);

  // 2) projections via WMMA: strips = (4096/16)*(256/64) = 1024 waves -> 128 blocks
  const int gemm_blocks = ((N_NODES / 16) * (HIDDEN / 64) + (TB / 32) - 1) / (TB / 32);
  gemm_wmma_bf16_kernel<<<gemm_blocks, TB, 0, stream>>>(xh, Wqh,  bq,  Q,  N_NODES, HIDDEN, HIDDEN, 0);
  gemm_wmma_bf16_kernel<<<gemm_blocks, TB, 0, stream>>>(xh, Wkh,  bk,  Kp, N_NODES, HIDDEN, HIDDEN, 0);
  gemm_wmma_bf16_kernel<<<gemm_blocks, TB, 0, stream>>>(xh, Wvh,  bv,  V,  N_NODES, HIDDEN, HIDDEN, 0);
  gemm_wmma_bf16_kernel<<<gemm_blocks, TB, 0, stream>>>(xh, Wgqh, bgq, Qg, N_NODES, HIDDEN, HIDDEN, 0);

  // 3) graph-level pieces
  colsum_kernel<<<1, TB, 0, stream>>>(x, ctx, N_NODES, HIDDEN, 1.0f / (float)N_NODES);
  colsum_kernel<<<1, TB, 0, stream>>>(V, sumV, N_NODES, HIDDEN, 1.0f);
  graph_vec_kernel<<<1, TB, 0, stream>>>(Wgk, bgk, Wgv, bgv, ctx, Kg, Vg);
  node_score_kernel<<<(N_NODES * 32) / TB, TB, 0, stream>>>(Qg, Kg, sraw, N_NODES);
  softmax_kernel<<<1, TB, 0, stream>>>(sraw, wgt, N_NODES);

  // 4) sparse edge attention (collapsed dense softmax)
  zero_f32_kernel<<<(N_NODES * HIDDEN + TB - 1) / TB, TB, 0, stream>>>(acc, N_NODES * HIDDEN);
  zero_f32_kernel<<<(N_NODES * HEADS + TB - 1) / TB, TB, 0, stream>>>(Z, N_NODES * HEADS);
  edge_scatter_kernel<<<N_EDGES / EDGES_PER_BLOCK, TB, 0, stream>>>(ei, Q, Kp, V, acc, Z);

  // 5) combine + fusion MLP (WMMA, K=512, bias + ReLU) -> d_out
  combine_kernel<<<N_NODES, TB, 0, stream>>>(acc, Z, sumV, wgt, Vg, comb);
  gemm_wmma_bf16_kernel<<<gemm_blocks, TB, 0, stream>>>(comb, Wfh, bf, out, N_NODES, HIDDEN, 2 * HIDDEN, 1);
}
